// PSABlock_26774826123897
// MI455X (gfx1250) — matrix operand order repaired and checked by reading, not compiled, not run on hardware
//
#include <hip/hip_runtime.h>
#include <cstdint>

// ---------------------------------------------------------------------------
// PSABlock for MI455X (gfx1250, wave32, WMMA).
// Activations layout: Act[n][k], n = b*4096 + h*64 + w  (NPOS=16384 rows),
//                                k = c*4 + qcomp        (K = 4*C).
// All 1x1 quaternion convs are GEMMs  D[n][m] = sum_k X[n][k] * Wham[m][k]
// executed with v_wmma_f32_16x16x32_bf16.
// ---------------------------------------------------------------------------

#define NPOS 16384   // B*H*W = 4*64*64
#define HW   4096    // H*W

typedef __attribute__((ext_vector_type(8)))  __bf16 v8bf;
typedef __attribute__((ext_vector_type(16))) __bf16 v16bf;
typedef __attribute__((ext_vector_type(8)))  float  v8f;

__device__ __forceinline__ uint16_t f2bf_bits(float f) {
  union { float f; uint32_t u; } v; v.f = f;
  uint32_t u = v.u;
  return (uint16_t)((u + 0x7FFFu + ((u >> 16) & 1u)) >> 16);  // RNE
}

// ------------------------- pack input x -> bf16 Act ------------------------
__global__ __launch_bounds__(256) void pack_x_kernel(
    const float* __restrict__ x, uint16_t* __restrict__ dst) {
  int idx = blockIdx.x * 256 + threadIdx.x;     // n*1024 + k, total 16.7M
  int n = idx >> 10, k = idx & 1023;
  int b = n >> 12, hw = n & 4095;
  dst[idx] = f2bf_bits(x[((long)(b * 1024 + k)) * HW + hw]);
}

__global__ __launch_bounds__(256) void pack_f32_kernel(
    const float* __restrict__ src, uint16_t* __restrict__ dst, int total) {
  int idx = blockIdx.x * 256 + threadIdx.x;
  if (idx < total) dst[idx] = f2bf_bits(src[idx]);
}

// ------------------- build Hamilton-structured weights ---------------------
// w: [4][O][I] f32 (1x1 conv).  out: bf16 [O*4][I*4] row-major.
__global__ __launch_bounds__(256) void build_ham_kernel(
    const float* __restrict__ w, uint16_t* __restrict__ out, int O, int I) {
  const int   widx[4][4] = {{0,1,2,3},{1,0,3,2},{2,3,0,1},{3,2,1,0}};
  const float sgn [4][4] = {{1,-1,-1,-1},{1,1,1,-1},{1,-1,1,1},{1,1,-1,1}};
  int idx = blockIdx.x * 256 + threadIdx.x;
  int Kq = I * 4;
  if (idx >= O * 4 * Kq) return;
  int m = idx / Kq, kk = idx - m * Kq;
  int co = m >> 2, qo = m & 3, ci = kk >> 2, qi = kk & 3;
  float v = sgn[qo][qi] * w[((long)widx[qo][qi] * O + co) * I + ci];
  out[idx] = f2bf_bits(v);
}

// ------------------------------ WMMA GEMM ----------------------------------
// D[n][m] = sum_k X[n][k] * W[m][k]; W:[M][K] bf16, X:[NPOS][K] bf16, D f32.
// Wave tile 32x32 (2x2 WMMA), block tile 64(m) x 128(n), 8 waves/block.
__global__ __launch_bounds__(256) void gemm_bf16_wmma(
    const uint16_t* __restrict__ Wq, const uint16_t* __restrict__ Xq,
    float* __restrict__ D, int M, int K) {
  int wave = threadIdx.x >> 5, lane = threadIdx.x & 31;
  int lm = lane & 15, hi = lane >> 4;
  int wm = wave >> 2, wn = wave & 3;
  long tile_n = (long)blockIdx.x * 128 + wn * 32;
  long tile_m = (long)blockIdx.y * 64  + wm * 32;

  const __bf16* Wb = (const __bf16*)Wq;
  const __bf16* Xb = (const __bf16*)Xq;
  const __bf16* Wr0 = Wb + (tile_m + lm) * (long)K;
  const __bf16* Wr1 = Wr0 + 16 * (long)K;
  const __bf16* Xr0 = Xb + (tile_n + lm) * (long)K;
  const __bf16* Xr1 = Xr0 + 16 * (long)K;

  v8f acc00 = {}, acc01 = {}, acc10 = {}, acc11 = {};
  const int aoff = hi * 16, boff = hi * 16;

  for (int k0 = 0; k0 < K; k0 += 32) {
    __builtin_prefetch((const void*)(Wr0 + k0 + 128), 0, 0);  // global_prefetch
    __builtin_prefetch((const void*)(Xr0 + k0 + 128), 0, 0);
    v8bf a0l = *(const v8bf*)(Wr0 + k0 + aoff);
    v8bf a0h = *(const v8bf*)(Wr0 + k0 + aoff + 8);
    v8bf a1l = *(const v8bf*)(Wr1 + k0 + aoff);
    v8bf a1h = *(const v8bf*)(Wr1 + k0 + aoff + 8);
    v8bf b0l = *(const v8bf*)(Xr0 + k0 + boff);
    v8bf b0h = *(const v8bf*)(Xr0 + k0 + boff + 8);
    v8bf b1l = *(const v8bf*)(Xr1 + k0 + boff);
    v8bf b1h = *(const v8bf*)(Xr1 + k0 + boff + 8);
    v16bf A0 = __builtin_shufflevector(a0l, a0h, 0,1,2,3,4,5,6,7,8,9,10,11,12,13,14,15);
    v16bf A1 = __builtin_shufflevector(a1l, a1h, 0,1,2,3,4,5,6,7,8,9,10,11,12,13,14,15);
    v16bf B0 = __builtin_shufflevector(b0l, b0h, 0,1,2,3,4,5,6,7,8,9,10,11,12,13,14,15);
    v16bf B1 = __builtin_shufflevector(b1l, b1h, 0,1,2,3,4,5,6,7,8,9,10,11,12,13,14,15);
    acc00 = __builtin_amdgcn_wmma_f32_16x16x32_bf16(false, A0, false, B0, (short)0, acc00, false, false);
    acc01 = __builtin_amdgcn_wmma_f32_16x16x32_bf16(false, A0, false, B1, (short)0, acc01, false, false);
    acc10 = __builtin_amdgcn_wmma_f32_16x16x32_bf16(false, A1, false, B0, (short)0, acc10, false, false);
    acc11 = __builtin_amdgcn_wmma_f32_16x16x32_bf16(false, A1, false, B1, (short)0, acc11, false, false);
  }

  long mb0 = tile_m + hi * 8, mb1 = tile_m + 16 + hi * 8;
  float* p;
  p = D + (tile_n + lm) * (long)M + mb0;
  *(float4*)p       = make_float4(acc00[0], acc00[1], acc00[2], acc00[3]);
  *(float4*)(p + 4) = make_float4(acc00[4], acc00[5], acc00[6], acc00[7]);
  p = D + (tile_n + 16 + lm) * (long)M + mb0;
  *(float4*)p       = make_float4(acc01[0], acc01[1], acc01[2], acc01[3]);
  *(float4*)(p + 4) = make_float4(acc01[4], acc01[5], acc01[6], acc01[7]);
  p = D + (tile_n + lm) * (long)M + mb1;
  *(float4*)p       = make_float4(acc10[0], acc10[1], acc10[2], acc10[3]);
  *(float4*)(p + 4) = make_float4(acc10[4], acc10[5], acc10[6], acc10[7]);
  p = D + (tile_n + 16 + lm) * (long)M + mb1;
  *(float4*)p       = make_float4(acc11[0], acc11[1], acc11[2], acc11[3]);
  *(float4*)(p + 4) = make_float4(acc11[4], acc11[5], acc11[6], acc11[7]);
}

// ----------------------------- attention -----------------------------------
// qkv: f32 [n][3072], m = (head*96 + idx)*4 + qc;  idx<32 q, 32..63 k, 64..95 v
// oattn: f32 [n][1024], k = (head*32 + c)*4 + qc
__global__ __launch_bounds__(256) void attention_kernel(
    const float* __restrict__ qkv, float* __restrict__ oattn) {
  const float SCALE = 0.17677669529663687f;  // 32^-0.5
  int blk = blockIdx.x;                       // 128 = 4b * 8h * 4qc
  int b = blk >> 5, head = (blk >> 2) & 7, qc = blk & 3;
  __shared__ float S[32][33];
  int t = threadIdx.x;
  int d = t & 31, cb = t >> 5;                // cb: 0..7
  long nbase = (long)b * HW;

  float acc[4] = {0.f, 0.f, 0.f, 0.f};
  int mk = (head * 96 + 32 + d) * 4 + qc;
  for (int n = 0; n < HW; n++) {
    const float* row = qkv + (nbase + n) * 3072;
    float kv = row[mk];
#pragma unroll
    for (int j = 0; j < 4; j++)
      acc[j] += row[(head * 96 + cb + 8 * j) * 4 + qc] * kv;
  }
#pragma unroll
  for (int j = 0; j < 4; j++) S[cb + 8 * j][d] = acc[j] * SCALE;
  __syncthreads();

  if (t < 32) {  // softmax over d for row c = t
    float mx = -3.4e38f;
    for (int dd = 0; dd < 32; dd++) mx = fmaxf(mx, S[t][dd]);
    float e[32], sum = 0.f;
    for (int dd = 0; dd < 32; dd++) { e[dd] = __expf(S[t][dd] - mx); sum += e[dd]; }
    float inv = 1.f / sum;
    for (int dd = 0; dd < 32; dd++) S[t][dd] = e[dd] * inv;
  }
  __syncthreads();

  for (int n = t; n < HW; n += 256) {
    const float* row = qkv + (nbase + n) * 3072;
    float v[32];
#pragma unroll
    for (int dd = 0; dd < 32; dd++) v[dd] = row[(head * 96 + 64 + dd) * 4 + qc];
    float* orow = oattn + (nbase + n) * 1024;
    for (int c = 0; c < 32; c++) {
      float a = 0.f;
#pragma unroll
      for (int dd = 0; dd < 32; dd++) a += S[c][dd] * v[dd];
      orow[(head * 32 + c) * 4 + qc] = a;
    }
  }
}

// ----------------- grouped 3x3 quaternion PE conv (+residual) --------------
// o: Act f32 [n][1024]; wpe: [4][256][4][3][3]; out = o + qconv(o, wpe, g=64)
__global__ __launch_bounds__(256) void pe_conv_kernel(
    const float* __restrict__ o, const float* __restrict__ wpe,
    float* __restrict__ out) {
  int n = blockIdx.x, oc = threadIdx.x;
  int hw = n & 4095, h = hw >> 6, w = hw & 63;
  long base = (long)n * 1024;
  float acc[4];
#pragma unroll
  for (int q = 0; q < 4; q++) acc[q] = o[base + oc * 4 + q];
  int g = oc >> 2;
  for (int dy = -1; dy <= 1; dy++) {
    int hh = h + dy; if (hh < 0 || hh >= 64) continue;
    for (int dx = -1; dx <= 1; dx++) {
      int ww = w + dx; if (ww < 0 || ww >= 64) continue;
      long nb = ((long)(n - hw) + hh * 64 + ww) * 1024;
      int tap = (dy + 1) * 3 + (dx + 1);
#pragma unroll
      for (int icl = 0; icl < 4; icl++) {
        int ic = g * 4 + icl;
        float xv[4], wv[4];
#pragma unroll
        for (int qi = 0; qi < 4; qi++) xv[qi] = o[nb + ic * 4 + qi];
#pragma unroll
        for (int i = 0; i < 4; i++)
          wv[i] = wpe[(((long)i * 256 + oc) * 4 + icl) * 9 + tap];
        acc[0] += wv[0]*xv[0] - wv[1]*xv[1] - wv[2]*xv[2] - wv[3]*xv[3];
        acc[1] += wv[1]*xv[0] + wv[0]*xv[1] + wv[3]*xv[2] - wv[2]*xv[3];
        acc[2] += wv[2]*xv[0] - wv[3]*xv[1] + wv[0]*xv[2] + wv[1]*xv[3];
        acc[3] += wv[3]*xv[0] + wv[2]*xv[1] - wv[1]*xv[2] + wv[0]*xv[3];
      }
    }
  }
#pragma unroll
  for (int q = 0; q < 4; q++) out[base + oc * 4 + q] = acc[q];
}

// --------------------------- IQBN column stats -----------------------------
__global__ __launch_bounds__(256) void zero_stats_kernel(float* s, int n) {
  int i = blockIdx.x * 256 + threadIdx.x;
  if (i < n) s[i] = 0.f;
}

__global__ __launch_bounds__(256) void colstats_kernel(
    const float* __restrict__ D, float* __restrict__ sums,
    float* __restrict__ sumsq, int K) {
  int k = blockIdx.x * 256 + threadIdx.x;
  int r0 = blockIdx.y * (NPOS / 16);
  float s = 0.f, s2 = 0.f;
  for (int n = r0; n < r0 + NPOS / 16; n++) {
    float v = D[(long)n * K + k];
    s += v; s2 += v * v;
  }
  atomicAdd(&sums[k], s);
  atomicAdd(&sumsq[k], s2);
}

// ----- X2 = x + iqbn(O3); also emit bf16 pack of X2 (FFN input) ------------
__global__ __launch_bounds__(256) void norm_add_proj_kernel(
    const float* __restrict__ O3, const float* __restrict__ x,
    const float* __restrict__ gam, const float* __restrict__ bet,
    const float* __restrict__ sums, const float* __restrict__ sumsq,
    float* __restrict__ X2, uint16_t* __restrict__ pack) {
  int idx = blockIdx.x * 256 + threadIdx.x;
  int n = idx >> 10, k = idx & 1023;
  float mean = sums[k] * (1.f / NPOS);
  float var  = sumsq[k] * (1.f / NPOS) - mean * mean;
  float v = gam[k] * (O3[idx] - mean) * rsqrtf(var + 1e-5f) + bet[k];
  int b = n >> 12, hw = n & 4095;
  float r = x[((long)(b * 1024 + k)) * HW + hw] + v;
  X2[idx] = r;
  pack[idx] = f2bf_bits(r);
}

// ----- relu(iqbn(F1)) -> bf16 pack (K=2048) --------------------------------
__global__ __launch_bounds__(256) void norm_relu_pack_kernel(
    const float* __restrict__ F1, const float* __restrict__ gam,
    const float* __restrict__ bet, const float* __restrict__ sums,
    const float* __restrict__ sumsq, uint16_t* __restrict__ pack) {
  int idx = blockIdx.x * 256 + threadIdx.x;
  int k = idx & 2047;
  float mean = sums[k] * (1.f / NPOS);
  float var  = sumsq[k] * (1.f / NPOS) - mean * mean;
  float v = gam[k] * (F1[idx] - mean) * rsqrtf(var + 1e-5f) + bet[k];
  pack[idx] = f2bf_bits(fmaxf(v, 0.f));
}

// ----- out = X2 + iqbn(F2), scattered back to [B,C,4,H,W] ------------------
__global__ __launch_bounds__(256) void final_kernel(
    const float* __restrict__ F2, const float* __restrict__ X2,
    const float* __restrict__ gam, const float* __restrict__ bet,
    const float* __restrict__ sums, const float* __restrict__ sumsq,
    float* __restrict__ out) {
  int idx = blockIdx.x * 256 + threadIdx.x;
  int n = idx >> 10, k = idx & 1023;
  float mean = sums[k] * (1.f / NPOS);
  float var  = sumsq[k] * (1.f / NPOS) - mean * mean;
  float v = gam[k] * (F2[idx] - mean) * rsqrtf(var + 1e-5f) + bet[k];
  int b = n >> 12, hw = n & 4095;
  out[((long)(b * 1024 + k)) * HW + hw] = X2[idx] + v;
}

// ---------------------------------------------------------------------------
extern "C" void kernel_launch(void* const* d_in, const int* in_sizes, int n_in,
                              void* d_out, int out_size, void* d_ws, size_t ws_size,
                              hipStream_t stream) {
  const float* x      = (const float*)d_in[0];
  const float* w_qkv  = (const float*)d_in[1];
  const float* w_proj = (const float*)d_in[2];
  const float* w_pe   = (const float*)d_in[3];
  const float* g_n    = (const float*)d_in[4];
  const float* b_n    = (const float*)d_in[5];
  const float* w_f1   = (const float*)d_in[6];
  const float* g_f1   = (const float*)d_in[7];
  const float* b_f1   = (const float*)d_in[8];
  const float* w_f2   = (const float*)d_in[9];
  const float* g_f2   = (const float*)d_in[10];
  const float* b_f2   = (const float*)d_in[11];

  char* ws = (char*)d_ws;
  float*    bufA  = (float*)(ws);                         // 192MB: QKV / O3 / F1
  float*    oattn = (float*)(ws + (size_t)201326592);     //  64MB
  float*    bufC  = (float*)(ws + (size_t)268435456);     //  64MB: O2 / F2
  float*    x2    = (float*)(ws + (size_t)335544320);     //  64MB
  uint16_t* packb = (uint16_t*)(ws + (size_t)402653184);  //  64MB bf16 pack
  uint16_t* wham  = (uint16_t*)(ws + (size_t)469762048);  //   6MB Hamilton W
  float*    sums  = (float*)(ws + (size_t)476053504);     //  16KB stats
  float*    sumsq = sums + 2048;

  // 1) pack x -> bf16 Act
  pack_x_kernel<<<65536, 256, 0, stream>>>(x, packb);
  // 2-3) QKV = GEMM (M=3072, K=1024)
  build_ham_kernel<<<12288, 256, 0, stream>>>(w_qkv, wham, 768, 256);
  gemm_bf16_wmma<<<dim3(128, 48), 256, 0, stream>>>(wham, packb, bufA, 3072, 1024);
  // 4) attention -> oattn
  attention_kernel<<<128, 256, 0, stream>>>(bufA, oattn);
  // 5) o2 = o + pe_conv(o)
  pe_conv_kernel<<<16384, 256, 0, stream>>>(oattn, w_pe, bufC);
  // 6-8) proj GEMM (M=1024, K=1024)
  pack_f32_kernel<<<65536, 256, 0, stream>>>(bufC, packb, NPOS * 1024);
  build_ham_kernel<<<4096, 256, 0, stream>>>(w_proj, wham, 256, 256);
  gemm_bf16_wmma<<<dim3(128, 16), 256, 0, stream>>>(wham, packb, bufA, 1024, 1024);
  // 9-10) iqbn + residual -> X2 (f32 + bf16 pack)
  zero_stats_kernel<<<16, 256, 0, stream>>>(sums, 4096);
  colstats_kernel<<<dim3(4, 16), 256, 0, stream>>>(bufA, sums, sumsq, 1024);
  norm_add_proj_kernel<<<65536, 256, 0, stream>>>(bufA, x, g_n, b_n, sums, sumsq, x2, packb);
  // 11-12) FFN1 GEMM (M=2048, K=1024)
  build_ham_kernel<<<8192, 256, 0, stream>>>(w_f1, wham, 512, 256);
  gemm_bf16_wmma<<<dim3(128, 32), 256, 0, stream>>>(wham, packb, bufA, 2048, 1024);
  // 13-14) relu(iqbn) -> bf16 pack (K=2048)
  zero_stats_kernel<<<16, 256, 0, stream>>>(sums, 4096);
  colstats_kernel<<<dim3(8, 16), 256, 0, stream>>>(bufA, sums, sumsq, 2048);
  norm_relu_pack_kernel<<<131072, 256, 0, stream>>>(bufA, g_f1, b_f1, sums, sumsq, packb);
  // 15-16) FFN2 GEMM (M=1024, K=2048)
  build_ham_kernel<<<8192, 256, 0, stream>>>(w_f2, wham, 256, 512);
  gemm_bf16_wmma<<<dim3(128, 16), 256, 0, stream>>>(wham, packb, bufC, 1024, 2048);
  // 17-18) out = X2 + iqbn(F2)
  zero_stats_kernel<<<16, 256, 0, stream>>>(sums, 4096);
  colstats_kernel<<<dim3(4, 16), 256, 0, stream>>>(bufC, sums, sumsq, 1024);
  final_kernel<<<65536, 256, 0, stream>>>(bufC, x2, g_f2, b_f2, sums, sumsq, (float*)d_out);
}
